// LePEAttention_25580825215550
// MI455X (gfx1250) — matrix operand order repaired and checked by reading, not compiled, not run on hardware
//
#include <hip/hip_runtime.h>
#include <hip/hip_bf16.h>

typedef __attribute__((ext_vector_type(16))) _Float16 v16h;
typedef __attribute__((ext_vector_type(8)))  _Float16 v8h;
typedef __attribute__((ext_vector_type(4)))  _Float16 v4h;
typedef __attribute__((ext_vector_type(8)))  float    v8f;

#define WIN   512
#define HD    32
#define HEADS 8
#define DIMC  256
#define NWIN  32             // 4 batches * 8 window strips
#define ATTN_PER 262144      // 512*512
#define X_ELEMS 4194304u     // 4*4096*256
#define VTP   528            // V^T LDS pitch (halves), 32B-aligned rows
#define SSP   520            // S-block LDS pitch (halves), bank-spread

// Dynamic LDS layout (halves):
//   Qs : 512*32            = 16384
//   Ks : 512*32            = 16384
//   Vt : 32*VTP            = 16896
//   Ss : 8 waves*16*SSP    = 66560
//   invS: 128 floats       (=256 halves)
#define SH_HALVES (16384 + 16384 + 16896 + 66560 + 256)
#define SH_BYTES  (SH_HALVES * 2)

// -------------------------------------------------------------------------
// Fused kernel: S = scale*QK^T -> row softmax -> X = P@V, one WG (8 waves)
// per (window, head).  attn written exactly once, never re-read from HBM.
// -------------------------------------------------------------------------
__global__ __launch_bounds__(256) void fused_attn_kernel(
    const float* __restrict__ qkv, float* __restrict__ attn,
    float* __restrict__ xout) {
  extern __shared__ _Float16 smem[];
  _Float16* Qs = smem;                 // [p][d] f16, Q pre-scaled
  _Float16* Ks = Qs + 16384;           // [p][d] f16
  _Float16* Vt = Ks + 16384;           // [d][p] f16 (V transposed)
  _Float16* Ss = Vt + 16896;           // per-wave [16][SSP] S blocks
  float*  invS = (float*)(Ss + 66560); // per-wave [16] row reciprocals

  const int bid = blockIdx.x;          // = w*8 + h
  const int w = bid >> 3, h = bid & 7;
  const int b = w >> 3, wx = w & 7;
  const int tid = threadIdx.x;
  const float SCALE = 0.17677669529663687f;   // 1/sqrt(32)
  const float* qg = qkv;
  const float* kg = qkv + (size_t)X_ELEMS;
  const float* vg = qkv + 2u * (size_t)X_ELEMS;

  // ---- Stage Q (scaled), K row-major and V^T as f16 ----
  #pragma unroll 4
  for (int t = 0; t < 64; ++t) {
    int e = tid + 256 * t;
    int p = e >> 5, d = e & 31;
    int L = (p >> 3) * 64 + wx * 8 + (p & 7);
    size_t gi = ((size_t)b * 4096 + L) * DIMC + h * HD + d;
    Qs[p * HD + d] = (_Float16)(qg[gi] * SCALE);
    Ks[p * HD + d] = (_Float16)(kg[gi]);
    Vt[d * VTP + p] = (_Float16)(vg[gi]);
  }
  __syncthreads();

  const int wv = tid >> 5, lane = tid & 31;
  const int n = lane & 15, hk = lane >> 4;
  _Float16* Sw = Ss + wv * 16 * SSP;   // this wave's 16x512 block
  float*  invw = invS + wv * 16;
  float* attn_base = attn + (size_t)bid * ATTN_PER;

  for (int rt = 0; rt < 4; ++rt) {
    const int rowbase = wv * 64 + rt * 16;

    // ---- S = Q K^T for 16 rows x 512 cols, into LDS (f16) ----
    v8h alo = *(const v8h*)(Qs + (rowbase + n) * HD + hk * 16);
    v8h ahi = *(const v8h*)(Qs + (rowbase + n) * HD + hk * 16 + 8);
    v16h afrag = __builtin_shufflevector(alo, ahi,
        0,1,2,3,4,5,6,7,8,9,10,11,12,13,14,15);
    for (int ct = 0; ct < 32; ++ct) {
      v16h bfrag = *(const v16h*)(Ks + (ct * 16 + n) * HD + hk * 16);
      v8f acc = {};
      acc = __builtin_amdgcn_wmma_f32_16x16x32_f16(
          false, afrag, false, bfrag, (short)0, acc, false, false);
      #pragma unroll
      for (int vv = 0; vv < 8; ++vv)
        Sw[(vv + 8 * hk) * SSP + ct * 16 + n] = (_Float16)acc[vv];
    }

    // ---- Row softmax: 2 lanes per row (lane n owns row n, halves by hk) --
    {
      const _Float16* srow = Sw + n * SSP + hk * 256;
      float mx = -3.4e38f;
      #pragma unroll 8
      for (int t = 0; t < 32; ++t) {
        v8h c8 = *(const v8h*)(srow + t * 8);
        #pragma unroll
        for (int u = 0; u < 8; ++u) mx = fmaxf(mx, (float)c8[u]);
      }
      mx = fmaxf(mx, __shfl_xor(mx, 16, 32));
      float s = 0.f;
      _Float16* srw = Sw + n * SSP + hk * 256;
      #pragma unroll 8
      for (int t = 0; t < 32; ++t) {
        v8h c8 = *(v8h*)(srw + t * 8);
        #pragma unroll
        for (int u = 0; u < 8; ++u) {
          float e = __expf((float)c8[u] - mx);
          s += e;
          c8[u] = (_Float16)e;
        }
        *(v8h*)(srw + t * 8) = c8;
      }
      s += __shfl_xor(s, 16, 32);
      if (hk == 0) invw[n] = 1.0f / s;
    }

    // ---- Normalize block in LDS + coalesced f32 attn write (only touch) --
    for (int r2 = 0; r2 < 16; ++r2) {
      const float invr = invw[r2];
      _Float16* sr = Sw + r2 * SSP;
      float* ar = attn_base + (size_t)(rowbase + r2) * WIN;
      #pragma unroll
      for (int t = 0; t < 4; ++t) {
        int col = (lane + 32 * t) * 4;
        v4h hv = *(const v4h*)(sr + col);
        float4 p;
        p.x = (float)hv[0] * invr; p.y = (float)hv[1] * invr;
        p.z = (float)hv[2] * invr; p.w = (float)hv[3] * invr;
        *(float4*)(ar + col) = p;
        hv[0] = (_Float16)p.x; hv[1] = (_Float16)p.y;
        hv[2] = (_Float16)p.z; hv[3] = (_Float16)p.w;
        *(v4h*)(sr + col) = hv;
      }
    }

    // ---- X = P @ V for this 16-row block (P from LDS, V^T from LDS) ----
    v8f acc0 = {}, acc1 = {};
    for (int kt = 0; kt < 16; ++kt) {
      v8h plo = *(const v8h*)(Sw + n * SSP + kt * 32 + hk * 16);
      v8h phi = *(const v8h*)(Sw + n * SSP + kt * 32 + hk * 16 + 8);
      v16h pfrag = __builtin_shufflevector(plo, phi,
          0,1,2,3,4,5,6,7,8,9,10,11,12,13,14,15);
      v16h b0 = *(const v16h*)(Vt + n * VTP + kt * 32 + hk * 16);
      v16h b1 = *(const v16h*)(Vt + (16 + n) * VTP + kt * 32 + hk * 16);
      acc0 = __builtin_amdgcn_wmma_f32_16x16x32_f16(
          false, pfrag, false, b0, (short)0, acc0, false, false);
      acc1 = __builtin_amdgcn_wmma_f32_16x16x32_f16(
          false, pfrag, false, b1, (short)0, acc1, false, false);
    }
    const int qs = rowbase + 8 * hk;
    #pragma unroll
    for (int vv = 0; vv < 8; ++vv) {
      int qq = qs + vv;
      int L = (qq >> 3) * 64 + wx * 8 + (qq & 7);
      size_t base = ((size_t)b * 4096 + L) * DIMC + h * HD;
      xout[base + n]      = acc0[vv];
      xout[base + 16 + n] = acc1[vv];
    }
  }
}

// -------------------------------------------------------------------------
// LePE depthwise 3x3 conv (zero pad inside each 64x8 window) + bias,
// accumulated into x. One thread per (b, L, c); c fastest (coalesced).
// -------------------------------------------------------------------------
__global__ __launch_bounds__(256) void lepe_kernel(
    const float* __restrict__ qkv, const float* __restrict__ cw,
    const float* __restrict__ cb, float* __restrict__ xout) {
  const size_t e = (size_t)blockIdx.x * 256 + threadIdx.x;
  const int c = (int)(e & 255);
  const int L = (int)((e >> 8) & 4095);
  const int b = (int)(e >> 20);
  const int i = L >> 6;
  const int col = L & 63;
  const int j = col & 7;
  const float* vg = qkv + 2u * (size_t)X_ELEMS;
  float acc = cb[c];
  #pragma unroll
  for (int di = -1; di <= 1; ++di) {
    int ii = i + di;
    if (ii < 0 || ii >= 64) continue;
    #pragma unroll
    for (int dj = -1; dj <= 1; ++dj) {
      int jj = j + dj;
      if (jj < 0 || jj >= 8) continue;      // zero pad at window edge
      int LL = ii * 64 + (col - j) + jj;
      acc += cw[c * 9 + (di + 1) * 3 + (dj + 1)] *
             vg[((size_t)b * 4096 + LL) * DIMC + c];
    }
  }
  xout[e] += acc;
}

// -------------------------------------------------------------------------
extern "C" void kernel_launch(void* const* d_in, const int* in_sizes, int n_in,
                              void* d_out, int out_size, void* d_ws, size_t ws_size,
                              hipStream_t stream) {
  const float* qkv    = (const float*)d_in[0];   // [3,4,4096,256] f32
  const float* conv_w = (const float*)d_in[1];   // [256,1,3,3]
  const float* conv_b = (const float*)d_in[2];   // [256]
  float* x    = (float*)d_out;                   // [4,4096,256]
  float* attn = x + (size_t)X_ELEMS;             // [32,8,512,512]

  fused_attn_kernel<<<NWIN * HEADS, 256, SH_BYTES, stream>>>(qkv, attn, x);
  lepe_kernel<<<X_ELEMS / 256, 256, 0, stream>>>(qkv, conv_w, conv_b, x);
}
